// K_OPT_5239860101687
// MI455X (gfx1250) — compile-verified, matrix operand order checked by reading, not run on hardware
//
#include <hip/hip_runtime.h>

typedef __attribute__((ext_vector_type(16))) _Float16 v16h;
typedef __attribute__((ext_vector_type(8)))  _Float16 v8h;
typedef __attribute__((ext_vector_type(8)))  float    v8f;
typedef __attribute__((ext_vector_type(2)))  float    v2f;

#define KDIM   256
#define TSTEPS 256
#define BATCH  1024
#define BM     16          // batch rows per workgroup (one WMMA M tile)
#define NWAVES 16          // one 16-column tile per wave
#define BLOCK  (NWAVES * 32)
#define DT     0.01f

// ---------------------------------------------------------------------------
// Pre-pass: W (f32, [k][n] row-major) -> Wt (f16, [n][k] row-major).
// A WMMA B-fragment column n then becomes a contiguous run of k values.
// ---------------------------------------------------------------------------
__global__ void wcvt_kernel(const float* __restrict__ W, _Float16* __restrict__ Wt) {
    const int n = blockIdx.x;
    const int k = threadIdx.x;
    Wt[n * KDIM + k] = (_Float16)W[k * KDIM + n];
}

// Load one 16x32 f16 fragment (A or B-column layout) from a row-major
// [16-rows x 256] f16 matrix: two contiguous 16B chunks per lane.
__device__ __forceinline__ v16h load_frag(const _Float16* base) {
    v16h f;
    const v8h* p = (const v8h*)base;
    ((v8h*)&f)[0] = p[0];   // K = kb .. kb+7
    ((v8h*)&f)[1] = p[2];   // K = kb+16 .. kb+23
    return f;
}

// ---------------------------------------------------------------------------
// Persistent Koopman rollout. 64 WGs x 512 threads (16 waves).
// Each WG owns BM=16 batch rows; wave w owns output-column tile w.
// Weights live in VGPRs for the whole rollout; A comes from LDS.
// ---------------------------------------------------------------------------
__global__ __launch_bounds__(BLOCK, 1) void koop_kernel(
    const float*    __restrict__ x,
    const _Float16* __restrict__ W1t,
    const _Float16* __restrict__ W2t,
    const float*    __restrict__ b1,
    const float*    __restrict__ b2,
    float*          __restrict__ out)
{
    __shared__ __align__(16) float    sy  [BM * KDIM];  // state, f32    (16 KB)
    __shared__ __align__(16) _Float16 sy16[BM * KDIM];  // state, f16    ( 8 KB)
    __shared__ __align__(16) _Float16 sh16[BM * KDIM];  // tanh(h), f16  ( 8 KB)

    const int tid   = threadIdx.x;
    const int wave  = tid >> 5;
    const int lane  = tid & 31;
    const int hr    = lane & 15;   // row (A) / column (B,C,D) inside the tile
    const int hi    = lane >> 4;   // lane-half: K sub-slice / rows M+8
    const int rowg0 = blockIdx.x * BM;
    const int c0    = wave * 16 + hr;        // this lane's output column
    const int colE  = c0 & ~1;               // even column of its (mu,omega) pair
    const int odd   = hr & 1;

    // ---- weights: resident in VGPRs for the entire rollout ---------------
    v16h B1[8], B2[8];
    #pragma unroll
    for (int kt = 0; kt < 8; ++kt) {
        const int kb = kt * 32 + hi * 8;
        B1[kt] = load_frag(W1t + c0 * KDIM + kb);
        B2[kt] = load_frag(W2t + c0 * KDIM + kb);
    }
    const float b1v = b1[c0];
    const float b2v = b2[c0];

    // ---- init y from x (x is (B,1,K) contiguous) --------------------------
    for (int i = tid; i < BM * KDIM; i += BLOCK) {
        float v = x[(size_t)rowg0 * KDIM + i];
        sy[i]   = v;
        sy16[i] = (_Float16)v;
    }

    const size_t out_row_stride = (size_t)TSTEPS * KDIM;
    float* outg = out + (size_t)rowg0 * out_row_stride;

    __syncthreads();

    for (int t = 0; t < TSTEPS; ++t) {
        // ========== GEMM1: h = tanh(y @ W1 + b1)  (y16 -> h16) =============
        {
            v16h A[8];
            #pragma unroll
            for (int kt = 0; kt < 8; ++kt)        // all A loads up front
                A[kt] = load_frag(sy16 + hr * KDIM + kt * 32 + hi * 8);
            v8f accA = {}, accB = {};             // 2 chains for MMA ILP
            #pragma unroll
            for (int kt = 0; kt < 8; kt += 2) {
                accA = __builtin_amdgcn_wmma_f32_16x16x32_f16(
                           false, A[kt],     false, B1[kt],     (short)0, accA, false, false);
                accB = __builtin_amdgcn_wmma_f32_16x16x32_f16(
                           false, A[kt + 1], false, B1[kt + 1], (short)0, accB, false, false);
            }
            const v8f acc = accA + accB;
            #pragma unroll
            for (int r = 0; r < 8; ++r) {
                const int row = r + hi * 8;       // C/D layout: M = r + 8*hi
                sh16[row * KDIM + c0] = (_Float16)tanhf(acc[r] + b1v);
            }
        }
        __syncthreads();

        // ========== GEMM2: p = h @ W2 + b2  (p stays in registers) =========
        v8f p;
        {
            v16h A[8];
            #pragma unroll
            for (int kt = 0; kt < 8; ++kt)
                A[kt] = load_frag(sh16 + hr * KDIM + kt * 32 + hi * 8);
            v8f accA = {}, accB = {};
            #pragma unroll
            for (int kt = 0; kt < 8; kt += 2) {
                accA = __builtin_amdgcn_wmma_f32_16x16x32_f16(
                           false, A[kt],     false, B2[kt],     (short)0, accA, false, false);
                accB = __builtin_amdgcn_wmma_f32_16x16x32_f16(
                           false, A[kt + 1], false, B2[kt + 1], (short)0, accB, false, false);
            }
            const v8f acc = accA + accB;
            #pragma unroll
            for (int r = 0; r < 8; ++r) p[r] = acc[r] + b2v;
        }

        // ========== rotation update: (mu,omega) pair via lane permute ======
        // Adjacent lanes hold adjacent columns -> shfl_xor(1) pairs them.
        #pragma unroll
        for (int r = 0; r < 8; ++r) {
            const int   row   = r + hi * 8;
            const float own   = p[r];
            const float other = __shfl_xor(own, 1, 32);
            const float mu    = odd ? other : own;
            const float om    = odd ? own   : other;
            const float e = __expf(DT * mu);
            const float c = __cosf(DT * om) * e;
            const float s = __sinf(DT * om) * e;
            const v2f  yp = *(const v2f*)(sy + row * KDIM + colE);   // (y0,y1)
            const float n = odd ? (s * yp.x + c * yp.y)    // new odd element
                                : (c * yp.x - s * yp.y);   // new even element
            sy  [row * KDIM + c0] = n;
            sy16[row * KDIM + c0] = (_Float16)n;
            outg[(size_t)row * out_row_stride + (size_t)t * KDIM + c0] = n;
        }
        __syncthreads();   // y/y16 visible before next step's GEMM1
    }
}

// ---------------------------------------------------------------------------
extern "C" void kernel_launch(void* const* d_in, const int* in_sizes, int n_in,
                              void* d_out, int out_size, void* d_ws, size_t ws_size,
                              hipStream_t stream) {
    const float* x  = (const float*)d_in[0];
    const float* W1 = (const float*)d_in[1];
    const float* b1 = (const float*)d_in[2];
    const float* W2 = (const float*)d_in[3];
    const float* b2 = (const float*)d_in[4];
    float* out = (float*)d_out;

    _Float16* W1t = (_Float16*)d_ws;                    // 128 KB
    _Float16* W2t = W1t + (size_t)KDIM * KDIM;          // 128 KB

    wcvt_kernel<<<KDIM, KDIM, 0, stream>>>(W1, W1t);
    wcvt_kernel<<<KDIM, KDIM, 0, stream>>>(W2, W2t);
    koop_kernel<<<BATCH / BM, BLOCK, 0, stream>>>(x, W1t, W2t, b1, b2, out);
}
